// GinAR_31653908972075
// MI455X (gfx1250) — compile-verified, matrix operand order checked by reading, not run on hardware
//
#include <hip/hip_runtime.h>
#include <hip/hip_bf16.h>
#include <stdint.h>

// ---------------- problem constants ----------------
#define B_   16
#define L_   96
#define N_   1024
#define C_   3
#define EMB_ 64
#define GRAP_ 16
#define OUT_ 12
#define NLAYER_ 2
#define BL_  (B_ * L_)            // 1536
#define RTOT_ ((size_t)BL_ * N_)  // 1,572,864 rows of EMB
#define BLNE_ (RTOT_ * EMB_)      // 100,663,296 elements

typedef __attribute__((ext_vector_type(16))) __bf16 bf16x16;
typedef __attribute__((ext_vector_type(8)))  __bf16 bf16x8;
typedef __attribute__((ext_vector_type(8)))  float  f32x8;

__device__ __forceinline__ float sigmoidf_(float x) { return 1.0f / (1.0f + __expf(-x)); }

// ---- fragment loads: every fragment is two contiguous 8xbf16 (16B) runs ----
// A (16x32, MxK) from row-major [ld = row stride in elements], K-contiguous.
__device__ __forceinline__ bf16x16 load_afrag(const __bf16* __restrict__ base, int ld, int k0, int lane) {
    const int r = lane & 15, kb = (lane >> 4) * 8;
    const __bf16* p = base + (size_t)r * ld + k0 + kb;
    bf16x16 a;
#pragma unroll
    for (int j = 0; j < 8; ++j) { a[j] = p[j]; a[8 + j] = p[16 + j]; }
    return a;
}
// B (32x16, KxN) from a TRANSPOSED buffer T[e][k] with row stride ldk (K-contiguous per col).
__device__ __forceinline__ bf16x16 load_bfrag_t(const __bf16* __restrict__ T, int ldk, int e0, int k0, int lane) {
    const int c = lane & 15, kb = (lane >> 4) * 8;
    const __bf16* p = T + (size_t)(e0 + c) * ldk + k0 + kb;
    bf16x16 b;
#pragma unroll
    for (int j = 0; j < 8; ++j) { b[j] = p[j]; b[8 + j] = p[16 + j]; }
    return b;
}
// B fragment from a pre-swizzled weight buffer: [mat][kk][nt][lane][16] contiguous.
__device__ __forceinline__ bf16x16 load_bfrag_swz(const __bf16* __restrict__ Wswz, int mat, int kk, int nt, int lane) {
    const __bf16* p = Wswz + ((size_t)(((mat * 2 + kk) * 4 + nt) * 32 + lane) << 4);
    bf16x16 b;
#pragma unroll
    for (int j = 0; j < 16; ++j) b[j] = p[j];
    return b;
}

// ---------------- small prep kernels ----------------
__global__ void k_f2bf(const float* __restrict__ src, __bf16* __restrict__ dst, size_t n) {
    size_t i = (size_t)blockIdx.x * blockDim.x + threadIdx.x;
    size_t stride = (size_t)gridDim.x * blockDim.x;
    for (; i < n; i += stride) dst[i] = (__bf16)src[i];
}

// Swizzle nmat 64x64 fp32 weights into WMMA-B fragment order (bf16):
// out[(((mat*2+kk)*4+nt)*32+lane)*16 + j] = W[mat][k][col],
//   k = kk*32 + (lane>>4)*8 + (j<8 ? j : 8+j), col = nt*16 + (lane&15)
__global__ void k_wswz(const float* __restrict__ W, __bf16* __restrict__ out, int nmat) {
    const int o = blockIdx.x * 256 + threadIdx.x;
    if (o >= nmat * 4096) return;
    const int j    = o & 15;
    const int lane = (o >> 4) & 31;
    const int nt   = (o >> 9) & 3;
    const int kk   = (o >> 11) & 1;
    const int mat  = o >> 12;
    const int jj   = (j < 8) ? j : (8 + j);           // 0..7 / 16..23
    const int k    = kk * 32 + ((lane >> 4) * 8) + jj;
    const int col  = nt * 16 + (lane & 15);
    out[o] = (__bf16)W[mat * 4096 + k * 64 + col];
}

// q = E@Wq, k = E@Wk  (1024x64 @ 64x64), one block per row m, 64 threads.
__global__ __launch_bounds__(64) void k_qk(const float* __restrict__ E,
                                           const float* __restrict__ Wq,
                                           const float* __restrict__ Wk,
                                           float* __restrict__ q, float* __restrict__ k) {
    const int m = blockIdx.x, e = threadIdx.x;
    __shared__ float Es[EMB_];
    Es[e] = E[m * EMB_ + e];
    __syncthreads();
    float aq = 0.f, ak = 0.f;
#pragma unroll 8
    for (int f = 0; f < EMB_; ++f) { aq += Es[f] * Wq[f * EMB_ + e]; ak += Es[f] * Wk[f * EMB_ + e]; }
    q[m * EMB_ + e] = aq;
    k[m * EMB_ + e] = ak;
}

// attn row: softmax over n of (q[m]·k[n]) * scale  -> bf16
__global__ __launch_bounds__(256) void k_attn(const float* __restrict__ q, const float* __restrict__ k,
                                              float scale, __bf16* __restrict__ out) {
    const int m = blockIdx.x, tid = threadIdx.x;
    __shared__ float qs[EMB_];
    __shared__ float s[N_];
    __shared__ float red[256];
    if (tid < EMB_) qs[tid] = q[m * EMB_ + tid];
    __syncthreads();
    float lmax = -3.0e38f;
    for (int n = tid; n < N_; n += 256) {
        float d = 0.f;
#pragma unroll 8
        for (int e = 0; e < EMB_; ++e) d += qs[e] * k[n * EMB_ + e];
        d *= scale;
        s[n] = d;
        lmax = fmaxf(lmax, d);
    }
    red[tid] = lmax; __syncthreads();
    for (int w = 128; w > 0; w >>= 1) { if (tid < w) red[tid] = fmaxf(red[tid], red[tid + w]); __syncthreads(); }
    const float gmax = red[0];
    __syncthreads();
    float lsum = 0.f;
    for (int n = tid; n < N_; n += 256) { float ev = __expf(s[n] - gmax); s[n] = ev; lsum += ev; }
    red[tid] = lsum; __syncthreads();
    for (int w = 128; w > 0; w >>= 1) { if (tid < w) red[tid] += red[tid + w]; __syncthreads(); }
    const float inv = 1.0f / red[0];
    for (int n = tid; n < N_; n += 256) out[(size_t)m * N_ + n] = (__bf16)(s[n] * inv);
}

// adp row: softmax over n of relu(Eg1[m]·Eg2[n]) -> bf16
__global__ __launch_bounds__(256) void k_adp(const float* __restrict__ Eg1, const float* __restrict__ Eg2,
                                             __bf16* __restrict__ out) {
    const int m = blockIdx.x, tid = threadIdx.x;
    __shared__ float g1[GRAP_];
    __shared__ float s[N_];
    __shared__ float red[256];
    if (tid < GRAP_) g1[tid] = Eg1[m * GRAP_ + tid];
    __syncthreads();
    float lmax = -3.0e38f;
    for (int n = tid; n < N_; n += 256) {
        float d = 0.f;
#pragma unroll
        for (int j = 0; j < GRAP_; ++j) d += g1[j] * Eg2[n * GRAP_ + j];
        d = fmaxf(d, 0.0f);
        s[n] = d;
        lmax = fmaxf(lmax, d);
    }
    red[tid] = lmax; __syncthreads();
    for (int w = 128; w > 0; w >>= 1) { if (tid < w) red[tid] = fmaxf(red[tid], red[tid + w]); __syncthreads(); }
    const float gmax = red[0];
    __syncthreads();
    float lsum = 0.f;
    for (int n = tid; n < N_; n += 256) { float ev = __expf(s[n] - gmax); s[n] = ev; lsum += ev; }
    red[tid] = lsum; __syncthreads();
    for (int w = 128; w > 0; w >>= 1) { if (tid < w) red[tid] += red[tid + w]; __syncthreads(); }
    const float inv = 1.0f / red[0];
    for (int n = tid; n < N_; n += 256) out[(size_t)m * N_ + n] = (__bf16)(s[n] * inv);
}

// Vt[bl][e][n] = (X @ Wv + bv)^T per slice, written TRANSPOSED with coalesced stores.
// grid: (BL, 16 n-tiles of 64), block 256.
__global__ __launch_bounds__(256) void k_v(const float* __restrict__ X, const float* __restrict__ Wv,
                                           const float* __restrict__ bvz, __bf16* __restrict__ Vt, int Cin) {
    __shared__ float Ws[EMB_ * EMB_];
    __shared__ float bs[EMB_];
    __shared__ float xs[64][EMB_];
    const int bl = blockIdx.x, ntile = blockIdx.y, tid = threadIdx.x;
    for (int i = tid; i < Cin * EMB_; i += 256) Ws[i] = Wv[i];
    if (tid < EMB_) bs[tid] = bvz[tid];
    const size_t rowbase = (size_t)bl * N_ + (size_t)ntile * 64;   // first of 64 x-rows
    for (int i = tid; i < 64 * Cin; i += 256) {
        int rr = i / Cin, cc = i % Cin;
        xs[rr][cc] = X[(rowbase + rr) * Cin + cc];
    }
    __syncthreads();
    const int nloc = tid & 63;         // column (node) within tile -> contiguous store dim
    const int eq = tid >> 6;           // 0..3
    __bf16* outb = Vt + (size_t)bl * ((size_t)EMB_ * N_) + (size_t)ntile * 64 + nloc;
#pragma unroll
    for (int i = 0; i < 16; ++i) {
        const int e = eq + 4 * i;
        float acc = bs[e];
        for (int c = 0; c < Cin; ++c) acc += xs[nloc][c] * Ws[c * EMB_ + e];
        outb[(size_t)e * N_] = (__bf16)acc;
    }
}

// ---------------- WMMA graph GEMM: out[bl] = S(1024x1024) @ Bt[bl]^T ----------------
// S row-major bf16 (A operand), Bt transposed (EMB x N per slice, e-major).
// Writes row-major out (N x EMB per slice); optionally also transposed outT (EMB x N).
__global__ __launch_bounds__(256) void k_graph_gemm(const __bf16* __restrict__ S,
                                                    const __bf16* __restrict__ Bt,
                                                    __bf16* __restrict__ Hout,
                                                    __bf16* __restrict__ HoutT) {
    const int bl = blockIdx.x;       // 0..BL-1
    const int mblk = blockIdx.y;     // 0..15
    const int tid = threadIdx.x;
    const int wave = tid >> 5, lane = tid & 31;
    const int mt = wave & 3;               // m sub-tile 0..3
    const int ntb = (wave >> 2) * 2;       // n sub-tile base {0,2}
    f32x8 acc0 = {}, acc1 = {};
    const __bf16* Abase = S + (size_t)(mblk * 64 + mt * 16) * N_;
    const __bf16* Tbl = Bt + (size_t)bl * ((size_t)EMB_ * N_);
    for (int k0 = 0; k0 < N_; k0 += 32) {
        bf16x16 a  = load_afrag(Abase, N_, k0, lane);
        bf16x16 b0 = load_bfrag_t(Tbl, N_, ntb * 16,      k0, lane);
        bf16x16 b1 = load_bfrag_t(Tbl, N_, ntb * 16 + 16, k0, lane);
        acc0 = __builtin_amdgcn_wmma_f32_16x16x32_bf16(false, a, false, b0, (short)0, acc0, false, false);
        acc1 = __builtin_amdgcn_wmma_f32_16x16x32_bf16(false, a, false, b1, (short)0, acc1, false, false);
    }
    // C/D layout: vgpr r, lane l -> m = r + 8*(l>=16), col = l&15
    const int col = lane & 15;
    const int mofs = (lane < 16) ? 0 : 8;
    const int mbase = mblk * 64 + mt * 16 + mofs;
    // row-major store (scattered b16; amortized over 32 k-steps)
    __bf16* out = Hout + (size_t)bl * ((size_t)N_ * EMB_) + (size_t)mbase * EMB_;
#pragma unroll
    for (int r8 = 0; r8 < 8; ++r8) {
        out[(size_t)r8 * EMB_ + ntb * 16 + col]      = (__bf16)acc0[r8];
        out[(size_t)r8 * EMB_ + ntb * 16 + 16 + col] = (__bf16)acc1[r8];
    }
    // transposed store: 8 consecutive m per lane -> one b128 per subtile
    if (HoutT) {
        bf16x8 p0, p1;
#pragma unroll
        for (int r8 = 0; r8 < 8; ++r8) { p0[r8] = (__bf16)acc0[r8]; p1[r8] = (__bf16)acc1[r8]; }
        __bf16* outt = HoutT + (size_t)bl * ((size_t)EMB_ * N_);
        *(bf16x8*)(outt + (size_t)(ntb * 16 + col) * N_ + mbase)        = p0;
        *(bf16x8*)(outt + (size_t)(ntb * 16 + 16 + col) * N_ + mbase)   = p1;
    }
}

// ---------------- HG = A1@Wm0 + A2@Wm1 + H@Wm2 + bm (WMMA, K=64) ----------------
// Weights pre-swizzled into fragment order; no LDS.
__global__ __launch_bounds__(256) void k_mix(const __bf16* __restrict__ H,
                                             const __bf16* __restrict__ A1,
                                             const __bf16* __restrict__ A2,
                                             const __bf16* __restrict__ WmSwz,  // 3 mats, frag order
                                             const float* __restrict__ bm,      // 64
                                             __bf16* __restrict__ HG) {
    const int tid = threadIdx.x;
    const int wave = tid >> 5, lane = tid & 31;
    const int mt = wave & 3, ntb = (wave >> 2) * 2;
    const size_t row0 = (size_t)blockIdx.x * 64 + (size_t)mt * 16;
    f32x8 acc0 = {}, acc1 = {};
    const __bf16* srcs[3] = { A1, A2, H };   // pairs with Wm0, Wm1, Wm2
#pragma unroll
    for (int s = 0; s < 3; ++s) {
        const __bf16* base = srcs[s] + row0 * EMB_;
#pragma unroll
        for (int kk = 0; kk < 2; ++kk) {
            bf16x16 a  = load_afrag(base, EMB_, kk * 32, lane);
            bf16x16 b0 = load_bfrag_swz(WmSwz, s, kk, ntb,     lane);
            bf16x16 b1 = load_bfrag_swz(WmSwz, s, kk, ntb + 1, lane);
            acc0 = __builtin_amdgcn_wmma_f32_16x16x32_bf16(false, a, false, b0, (short)0, acc0, false, false);
            acc1 = __builtin_amdgcn_wmma_f32_16x16x32_bf16(false, a, false, b1, (short)0, acc1, false, false);
        }
    }
    const int col = lane & 15;
    const int mofs = (lane < 16) ? 0 : 8;
    const float bc0 = bm[ntb * 16 + col], bc1 = bm[ntb * 16 + 16 + col];
    __bf16* out = HG + (row0 + mofs) * EMB_;
#pragma unroll
    for (int r8 = 0; r8 < 8; ++r8) {
        out[(size_t)r8 * EMB_ + ntb * 16 + col]      = (__bf16)(acc0[r8] + bc0);
        out[(size_t)r8 * EMB_ + ntb * 16 + 16 + col] = (__bf16)(acc1[r8] + bc1);
    }
}

// ---------------- gates[g] = HG @ Wg[g] + bg[g] (WMMA, K=64, 4 gates) ----------------
__global__ __launch_bounds__(256) void k_gates(const __bf16* __restrict__ HG,
                                               const __bf16* __restrict__ WgSwz, // 4 mats, frag order
                                               const float* __restrict__ bg4,    // 4*64
                                               __bf16* __restrict__ Gt) {
    const int tid = threadIdx.x;
    const int wave = tid >> 5, lane = tid & 31;
    const int mt = wave & 3, ntb = (wave >> 2) * 2;
    const size_t row0 = (size_t)blockIdx.x * 64 + (size_t)mt * 16;
    const __bf16* base = HG + row0 * EMB_;
    bf16x16 aLo = load_afrag(base, EMB_, 0, lane);
    bf16x16 aHi = load_afrag(base, EMB_, 32, lane);
    const int col = lane & 15;
    const int mofs = (lane < 16) ? 0 : 8;
#pragma unroll
    for (int g = 0; g < 4; ++g) {
        f32x8 acc0 = {}, acc1 = {};
        bf16x16 b;
        b = load_bfrag_swz(WgSwz, g, 0, ntb, lane);
        acc0 = __builtin_amdgcn_wmma_f32_16x16x32_bf16(false, aLo, false, b, (short)0, acc0, false, false);
        b = load_bfrag_swz(WgSwz, g, 1, ntb, lane);
        acc0 = __builtin_amdgcn_wmma_f32_16x16x32_bf16(false, aHi, false, b, (short)0, acc0, false, false);
        b = load_bfrag_swz(WgSwz, g, 0, ntb + 1, lane);
        acc1 = __builtin_amdgcn_wmma_f32_16x16x32_bf16(false, aLo, false, b, (short)0, acc1, false, false);
        b = load_bfrag_swz(WgSwz, g, 1, ntb + 1, lane);
        acc1 = __builtin_amdgcn_wmma_f32_16x16x32_bf16(false, aHi, false, b, (short)0, acc1, false, false);
        const float bc0 = bg4[g * EMB_ + ntb * 16 + col];
        const float bc1 = bg4[g * EMB_ + ntb * 16 + 16 + col];
        __bf16* out = Gt + (size_t)g * BLNE_ + (row0 + mofs) * EMB_;
#pragma unroll
        for (int r8 = 0; r8 < 8; ++r8) {
            out[(size_t)r8 * EMB_ + ntb * 16 + col]      = (__bf16)(acc0[r8] + bc0);
            out[(size_t)r8 * EMB_ + ntb * 16 + 16 + col] = (__bf16)(acc1[r8] + bc1);
        }
    }
}

// ---------------- elementwise LSTM scan over t ----------------
__global__ __launch_bounds__(256) void k_scan(const __bf16* __restrict__ Gt,
                                              float* __restrict__ xout,        // (B,L,N,EMB)
                                              float* __restrict__ finals,      // (B,NLAYER,N,EMB)
                                              int z) {
    const size_t idx = (size_t)blockIdx.x * 256 + threadIdx.x;   // < B*N*EMB
    const int e = (int)(idx & 63);
    const size_t bn = idx >> 6;
    const int n = (int)(bn & (N_ - 1));
    const int b = (int)(bn >> 10);
    float c = 0.f, h = 0.f;
    for (int t = 0; t < L_; ++t) {
        const size_t base = (((size_t)(b * L_ + t)) * N_ + n) * EMB_ + e;
        const float f  = sigmoidf_((float)Gt[0 * BLNE_ + base]);
        const float ii = sigmoidf_((float)Gt[1 * BLNE_ + base]);
        const float g  = tanhf((float)Gt[2 * BLNE_ + base]);
        const float o  = sigmoidf_((float)Gt[3 * BLNE_ + base]);
        c = f * c + ii * g;
        h = o * tanhf(c);
        xout[base] = h;
    }
    finals[(((size_t)b * NLAYER_ + z) * N_ + n) * EMB_ + e] = h;
}

// ---------------- decoder ----------------
__global__ __launch_bounds__(256) void k_dec(const float* __restrict__ finals,
                                             const float* __restrict__ Wdec,  // (OUT,NLAYER,EMB)
                                             const float* __restrict__ bdec,
                                             const float* __restrict__ Wout,  // (OUT,OUT)
                                             const float* __restrict__ bout,
                                             float* __restrict__ out) {       // (B,OUT,N)
    __shared__ float Wd[OUT_ * NLAYER_ * EMB_];
    __shared__ float Wo[OUT_ * OUT_];
    __shared__ float bd[OUT_];
    __shared__ float bo[OUT_];
    const int tid = threadIdx.x;
    for (int i = tid; i < OUT_ * NLAYER_ * EMB_; i += 256) Wd[i] = Wdec[i];
    for (int i = tid; i < OUT_ * OUT_; i += 256) Wo[i] = Wout[i];
    if (tid < OUT_) { bd[tid] = bdec[tid]; bo[tid] = bout[tid]; }
    __syncthreads();
    const int idx = blockIdx.x * 256 + tid;  // < B*N
    const int n = idx & (N_ - 1);
    const int b = idx >> 10;
    float dec[OUT_];
#pragma unroll
    for (int o = 0; o < OUT_; ++o) dec[o] = bd[o];
    for (int z = 0; z < NLAYER_; ++z) {
        const float* fp = finals + (((size_t)b * NLAYER_ + z) * N_ + n) * EMB_;
        for (int e = 0; e < EMB_; ++e) {
            const float fv = fp[e];
#pragma unroll
            for (int o = 0; o < OUT_; ++o) dec[o] += fv * Wd[(o * NLAYER_ + z) * EMB_ + e];
        }
    }
#pragma unroll
    for (int o = 0; o < OUT_; ++o) {
        float acc = bo[o];
#pragma unroll
        for (int p = 0; p < OUT_; ++p) acc += Wo[o * OUT_ + p] * dec[p];
        out[((size_t)b * OUT_ + o) * N_ + n] = acc;
    }
}

// ---------------- host launcher ----------------
extern "C" void kernel_launch(void* const* d_in, const int* in_sizes, int n_in,
                              void* d_out, int out_size, void* d_ws, size_t ws_size,
                              hipStream_t stream) {
    const float* history = (const float*)d_in[0];
    const float* adj     = (const float*)d_in[1];
    const float* Wv0     = (const float*)d_in[2];
    const float* Wv1     = (const float*)d_in[3];
    const float* bv      = (const float*)d_in[4];
    const float* E       = (const float*)d_in[5];
    const float* Wq      = (const float*)d_in[6];
    const float* Wk      = (const float*)d_in[7];
    const float* Eg1     = (const float*)d_in[8];
    const float* Eg2     = (const float*)d_in[9];
    const float* Wmix    = (const float*)d_in[10];
    const float* bmix    = (const float*)d_in[11];
    const float* Wgates  = (const float*)d_in[12];
    const float* bgates  = (const float*)d_in[13];
    const float* Wdec    = (const float*)d_in[14];
    const float* bdec    = (const float*)d_in[15];
    const float* Wout    = (const float*)d_in[16];
    const float* bout    = (const float*)d_in[17];

    char* ws = (char*)d_ws;
    size_t off = 0;
    auto alloc = [&](size_t bytes) -> void* {
        void* p = ws + off;
        off = (off + bytes + 255) & ~(size_t)255;
        return p;
    };
    const size_t NN = (size_t)N_ * N_;
    __bf16* attnB = (__bf16*)alloc(NN * 2);
    __bf16* adpB  = (__bf16*)alloc(NN * 2);
    __bf16* adjB  = (__bf16*)alloc(NN * 2);
    float*  qbuf  = (float*)alloc((size_t)N_ * EMB_ * 4);
    float*  kbuf  = (float*)alloc((size_t)N_ * EMB_ * 4);
    __bf16* WmSwz = (__bf16*)alloc((size_t)3 * EMB_ * EMB_ * 2);
    __bf16* WgSwz = (__bf16*)alloc((size_t)4 * EMB_ * EMB_ * 2);
    __bf16* VtB   = (__bf16*)alloc(BLNE_ * 2);   // transposed V; reused as HG later
    __bf16* Hb    = (__bf16*)alloc(BLNE_ * 2);   // H row-major
    __bf16* HtB   = (__bf16*)alloc(BLNE_ * 2);   // H transposed
    __bf16* A1b   = (__bf16*)alloc(BLNE_ * 2);
    __bf16* A2b   = (__bf16*)alloc(BLNE_ * 2);
    __bf16* Gt    = (__bf16*)alloc(4 * BLNE_ * 2);
    float*  xbuf  = (float*)alloc(BLNE_ * 4);
    float*  finals= (float*)alloc((size_t)B_ * NLAYER_ * N_ * EMB_ * 4);
    __bf16* HGb   = VtB;   // V dead once H/Ht exist
    (void)in_sizes; (void)n_in; (void)out_size; (void)ws_size;

    // adj -> bf16 once
    k_f2bf<<<1024, 256, 0, stream>>>(adj, adjB, NN);

    for (int z = 0; z < NLAYER_; ++z) {
        const float* Ez   = E      + (size_t)z * N_ * EMB_;
        const float* Wqz  = Wq     + (size_t)z * EMB_ * EMB_;
        const float* Wkz  = Wk     + (size_t)z * EMB_ * EMB_;
        const float* Eg1z = Eg1    + (size_t)z * N_ * GRAP_;
        const float* Eg2z = Eg2    + (size_t)z * N_ * GRAP_;
        const float* Wmz  = Wmix   + (size_t)z * 3 * EMB_ * EMB_;
        const float* bmz  = bmix   + (size_t)z * EMB_;
        const float* Wgz  = Wgates + (size_t)z * 4 * EMB_ * EMB_;
        const float* bgz  = bgates + (size_t)z * 4 * EMB_;
        const float* bvz  = bv     + (size_t)z * EMB_;

        // graph matrices
        k_qk  <<<N_, 64, 0, stream>>>(Ez, Wqz, Wkz, qbuf, kbuf);
        k_attn<<<N_, 256, 0, stream>>>(qbuf, kbuf, 0.125f, attnB);
        k_adp <<<N_, 256, 0, stream>>>(Eg1z, Eg2z, adpB);
        // small weights -> fragment-swizzled bf16
        k_wswz<<<48, 256, 0, stream>>>(Wmz, WmSwz, 3);
        k_wswz<<<64, 256, 0, stream>>>(Wgz, WgSwz, 4);

        // V projection (writes transposed Vt)
        const float* X = (z == 0) ? history : xbuf;
        const float* Wv = (z == 0) ? Wv0 : Wv1;
        const int Cin = (z == 0) ? C_ : EMB_;
        dim3 gv(BL_, N_ / 64);
        k_v<<<gv, 256, 0, stream>>>(X, Wv, bvz, VtB, Cin);

        // graph convolutions (heavy WMMA GEMMs, no LDS / barriers)
        dim3 gg(BL_, N_ / 64);
        k_graph_gemm<<<gg, 256, 0, stream>>>(attnB, VtB, Hb, HtB);      // H = attn @ V (both layouts)
        k_graph_gemm<<<gg, 256, 0, stream>>>(adjB,  HtB, A1b, nullptr); // adj @ H
        k_graph_gemm<<<gg, 256, 0, stream>>>(adpB,  HtB, A2b, nullptr); // adp @ H

        // mix + gates (WMMA K=64, swizzled weights straight from L2)
        k_mix  <<<(unsigned)(RTOT_ / 64), 256, 0, stream>>>(Hb, A1b, A2b, WmSwz, bmz, HGb);
        k_gates<<<(unsigned)(RTOT_ / 64), 256, 0, stream>>>(HGb, WgSwz, bgz, Gt);

        // LSTM scan -> next-layer input + finals
        k_scan<<<(unsigned)((size_t)B_ * N_ * EMB_ / 256), 256, 0, stream>>>(Gt, xbuf, finals, z);
    }

    // decoder -> d_out (B, OUT, N) fp32
    k_dec<<<(unsigned)((B_ * N_) / 256), 256, 0, stream>>>(finals, Wdec, bdec, Wout, bout, (float*)d_out);
}
